// SpikingConv1DLayer_11381663334462
// MI455X (gfx1250) — matrix operand order repaired and checked by reading, not compiled, not run on hardware
//
#include <hip/hip_runtime.h>
#include <hip/hip_bf16.h>

// SpikingConv1DLayer for MI455X (gfx1250), wave32 + WMMA f16 hi/lo split.
// B=64, CIN=64, COUT=128, K=5 (padded to 8), T=2048.
// Workspace requirement: ~67 MB.

typedef _Float16 half_t;
typedef _Float16 v16h __attribute__((ext_vector_type(16)));
typedef float    v8f  __attribute__((ext_vector_type(8)));

#define NB    64
#define CIN   64
#define COUT  128
#define TLEN  2048
#define RS    24         // LDS x row stride (f32): 16 + 7 halo, padded

static __device__ __forceinline__ v8f wmma16(v16h a, v16h b, v8f c) {
    return __builtin_amdgcn_wmma_f32_16x16x32_f16(false, a, false, b, (short)0, c, false, false);
}

// ---------------- prep: d = W W^T, inv_norm -------------------------------
__global__ __launch_bounds__(256) void k_prep_d(const float* __restrict__ w,
                                                float* __restrict__ d,
                                                float* __restrict__ inv_norm) {
    int p  = blockIdx.x * 256 + threadIdx.x;      // 0..16383
    int o1 = p >> 7, o2 = p & 127;
    const float* w1 = w + o1 * 320;
    const float* w2 = w + o2 * 320;
    float s = 0.f;
    for (int i = 0; i < 320; ++i) s += w1[i] * w2[i];
    d[p] = s;
    if (o1 == o2) inv_norm[o1] = 1.0f / (s + 1e-8f);
}

// ---------------- pack d into WMMA-B register order, f16 hi/lo ------------
// element id = ((nt*4 + kc)*32 + lane)*16 + i ; B[K=o_src][N=o_dst]
__global__ __launch_bounds__(256) void k_pack_dB(const float* __restrict__ d,
                                                 half_t* __restrict__ hi,
                                                 half_t* __restrict__ lo) {
    int id = blockIdx.x * 256 + threadIdx.x;      // 16384
    int i = id & 15, lane = (id >> 4) & 31, kc = (id >> 9) & 3, nt = id >> 11;
    int o_dst  = nt * 16 + (lane & 15);
    int Klocal = (i & 7) + ((i >= 8) ? 16 : 0) + ((lane >= 16) ? 8 : 0);
    int o_src  = kc * 32 + Klocal;
    float v = d[o_src * 128 + o_dst];
    half_t h = (half_t)v;
    hi[id] = h;
    lo[id] = (half_t)(v - (float)h);
}

// ---------------- pack w^T (im2col B for conv), f16 hi/lo -----------------
// element id = ((nt*16 + kc)*32 + lane)*16 + i ; B[K=q=(c*8+k)][N=o]
__global__ __launch_bounds__(256) void k_pack_wB(const float* __restrict__ w,
                                                 half_t* __restrict__ hi,
                                                 half_t* __restrict__ lo) {
    int id = blockIdx.x * 256 + threadIdx.x;      // 65536
    int i = id & 15, lane = (id >> 4) & 31, kc = (id >> 9) & 15, nt = id >> 13;
    int o      = nt * 16 + (lane & 15);
    int Klocal = (i & 7) + ((i >= 8) ? 16 : 0) + ((lane >= 16) ? 8 : 0);
    int q = kc * 32 + Klocal;
    int c = q >> 3, k = q & 7;
    float v = (k < 5) ? w[o * 320 + c * 5 + k] : 0.0f;
    half_t h = (half_t)v;
    hi[id] = h;
    lo[id] = (half_t)(v - (float)h);
}

// ---------------- conv1d as WMMA GEMM: y[t][b][o] -------------------------
// block = (batch b, t-chunk of 16); 8 waves, wave = 16-wide o tile.
// M = t-offset (16), N = o (16), K = q=(c,k) (512, 16 chunks of 32).
__global__ __launch_bounds__(256) void k_conv(const float* __restrict__ x,
                                              const half_t* __restrict__ wBhi,
                                              const half_t* __restrict__ wBlo,
                                              float* __restrict__ y) {
    __shared__ float xs[CIN * RS];
    int b  = blockIdx.x >> 7;
    int t0 = (blockIdx.x & 127) << 4;
    int tid = threadIdx.x;
    // stage x[b][c][t0-2 .. t0+20] (23 valid, zero-padded)
    for (int idx = tid; idx < CIN * RS; idx += 256) {
        int c = idx / RS, j = idx - c * RS;
        int t = t0 + j - 2;
        xs[idx] = (j < 23 && t >= 0 && t < TLEN) ? x[(b * CIN + c) * TLEN + t] : 0.0f;
    }
    __syncthreads();

    int wv = tid >> 5, lane = tid & 31;
    int m = lane & 15, hl = lane >> 4;
    v8f acc = {};
    for (int kc = 0; kc < 16; ++kc) {
        v16h Bh = *(const v16h*)(wBhi + (((wv * 16 + kc) * 32) + lane) * 16);
        v16h Bl = *(const v16h*)(wBlo + (((wv * 16 + kc) * 32) + lane) * 16);
        v16h Ah, Al;
#pragma unroll
        for (int i = 0; i < 16; ++i) {
            int v = i >> 1, h = i & 1;
            int Klocal = ((v & 3) * 2 + h) + ((v >= 4) ? 16 : 0) + (hl ? 8 : 0);
            int q = kc * 32 + Klocal;
            int c = q >> 3, k = q & 7;
            float xv = xs[c * RS + m + k];          // x[b][c][t0+m+k-2]
            half_t xh = (half_t)xv;
            Ah[i] = xh;
            Al[i] = (half_t)(xv - (float)xh);
        }
        acc = wmma16(Ah, Bh, acc);
        acc = wmma16(Al, Bh, acc);
        acc = wmma16(Ah, Bl, acc);
    }
    int o = wv * 16 + m;
#pragma unroll
    for (int j = 0; j < 8; ++j) {
        int toff = j + (hl ? 8 : 0);
        y[((t0 + toff) * NB + b) * COUT + o] = acc[j];
    }
}

// ---------------- sequential scan: mem in VGPRs, spk via LDS WMMA-A -------
// 4 blocks (16 batches each) x 8 waves (16 channels each).
// Ping-pong spike buffers -> single barrier/step; conv input prefetched
// one step ahead so the global L2 latency is off the critical path.
__global__ __launch_bounds__(256) void k_scan(const float* __restrict__ y,
                                              const half_t* __restrict__ dBhi,
                                              const half_t* __restrict__ dBlo,
                                              const float* __restrict__ inv_norm,
                                              const float* __restrict__ bvec,
                                              const float* __restrict__ beta_p,
                                              unsigned* __restrict__ spk_bits) {
    __shared__ __align__(32) half_t As[2][4 * 32 * 16];  // ping-pong WMMA-A spikes
    int tid = threadIdx.x, wv = tid >> 5, lane = tid & 31;
    int b0 = blockIdx.x * 16;
    int m = lane & 15, hl = lane >> 4;

    v16h Bh[4], Bl[4];
#pragma unroll
    for (int kc = 0; kc < 4; ++kc) {
        Bh[kc] = *(const v16h*)(dBhi + ((wv * 4 + kc) * 32 + lane) * 16);
        Bl[kc] = *(const v16h*)(dBlo + ((wv * 4 + kc) * 32 + lane) * 16);
    }
    float invn = inv_norm[wv * 16 + m];
    float bth  = bvec[wv * 16 + m];
    float beta = beta_p[0];
    float omb  = 1.0f - beta;
    v8f mem = {};
    unsigned bits[8] = {0, 0, 0, 0, 0, 0, 0, 0};

    // per-element offsets: conv/out linear offset and A-layout spike address
    int yOff[8];   // (b*COUT + o) within one t-plane
    int sAddr[8];  // index into As[*]
#pragma unroll
    for (int j = 0; j < 8; ++j) {
        int mm = j + (hl ? 8 : 0);          // batch-row within tile (A row)
        int o  = wv * 16 + m;               // this wave's channel
        yOff[j] = (b0 + mm) * COUT + o;
        int kc = o >> 5;
        int Kl = o & 31;
        int r  = Kl & 15;
        int dl = (r >= 8) ? 16 : 0;         // high-lane half for K%16 >= 8
        int rp = r & 7;
        int hb = rp & 1, vl = rp >> 1;
        int v  = vl + ((Kl >= 16) ? 4 : 0);
        sAddr[j] = (kc * 32 + mm + dl) * 16 + (2 * v + hb);
    }
    for (int i = tid; i < 4 * 32 * 16; i += 256) As[0][i] = (half_t)0;

    // prefetch conv input for t = 0
    float inp_n[8];
#pragma unroll
    for (int j = 0; j < 8; ++j) inp_n[j] = y[yOff[j]];
    __syncthreads();

    for (int t = 0; t < TLEN; ++t) {
        int rb = t & 1, wb = rb ^ 1;
        // consume prefetch, issue loads for t+1 (hidden behind WMMA+barrier)
        float inp_c[8];
#pragma unroll
        for (int j = 0; j < 8; ++j) inp_c[j] = inp_n[j];
        int tn = (t < TLEN - 1) ? t + 1 : t;
        const float* yp = y + tn * (NB * COUT);
#pragma unroll
        for (int j = 0; j < 8; ++j) inp_n[j] = yp[yOff[j]];

        // previous-step spikes as WMMA A operand (aligned ds_load_b128 x2)
        const half_t* Ab = As[rb];
        v16h A0 = *(const v16h*)(Ab + (0 * 32 + lane) * 16);
        v16h A1 = *(const v16h*)(Ab + (1 * 32 + lane) * 16);
        v16h A2 = *(const v16h*)(Ab + (2 * 32 + lane) * 16);
        v16h A3 = *(const v16h*)(Ab + (3 * 32 + lane) * 16);
        v8f r0 = {}, r1 = {}, r2 = {}, r3 = {};
        r0 = wmma16(A0, Bh[0], r0); r0 = wmma16(A0, Bl[0], r0);
        r1 = wmma16(A1, Bh[1], r1); r1 = wmma16(A1, Bl[1], r1);
        r2 = wmma16(A2, Bh[2], r2); r2 = wmma16(A2, Bl[2], r2);
        r3 = wmma16(A3, Bh[3], r3); r3 = wmma16(A3, Bl[3], r3);
        v8f rst = (r0 + r1) + (r2 + r3);

        half_t spkh[8];
#pragma unroll
        for (int j = 0; j < 8; ++j) {
            float mv = (mem[j] - rst[j]) * beta + inp_c[j] * omb;
            mem[j] = mv;
            float mthr = mv * invn - bth;
            unsigned s = (mthr > 0.0f) ? 1u : 0u;
            bits[j] |= s << (t & 31);
            spkh[j] = (half_t)(float)s;
        }
        half_t* Aw = As[wb];
#pragma unroll
        for (int j = 0; j < 8; ++j) Aw[sAddr[j]] = spkh[j];
        __syncthreads();   // orders write(t)->read(t+1) and read(t)->write(t+1)

        if ((t & 31) == 31) {
#pragma unroll
            for (int j = 0; j < 8; ++j) {
                spk_bits[yOff[j] * 64 + (t >> 5)] = bits[j];
                bits[j] = 0;
            }
        }
    }
}

// ---------------- expand bits -> f32 [B][O][T] + popcount loss ------------
__global__ __launch_bounds__(256) void k_expand(const unsigned* __restrict__ bits,
                                                float* __restrict__ out,
                                                unsigned* __restrict__ counter) {
    __shared__ unsigned red[256];
    int id = blockIdx.x * 256 + threadIdx.x;      // (b*128+o)*64 + t32
    unsigned v = bits[id];
    float4* o4 = (float4*)(out + (size_t)id * 32);
#pragma unroll
    for (int g = 0; g < 8; ++g) {
        float4 f;
        f.x = ((v >> (g * 4 + 0)) & 1u) ? 1.f : 0.f;
        f.y = ((v >> (g * 4 + 1)) & 1u) ? 1.f : 0.f;
        f.z = ((v >> (g * 4 + 2)) & 1u) ? 1.f : 0.f;
        f.w = ((v >> (g * 4 + 3)) & 1u) ? 1.f : 0.f;
        o4[g] = f;
    }
    red[threadIdx.x] = __popc(v);
    __syncthreads();
    for (int s = 128; s > 0; s >>= 1) {
        if (threadIdx.x < s) red[threadIdx.x] += red[threadIdx.x + s];
        __syncthreads();
    }
    if (threadIdx.x == 0) atomicAdd(counter, red[0]);
}

__global__ void k_zero(unsigned* counter) { *counter = 0u; }

__global__ void k_loss(const unsigned* __restrict__ counter, float* __restrict__ out) {
    out[16777216] = 0.5f * (float)(*counter) / 16777216.0f;
}

// --------------------------------------------------------------------------
extern "C" void kernel_launch(void* const* d_in, const int* in_sizes, int n_in,
                              void* d_out, int out_size, void* d_ws, size_t ws_size,
                              hipStream_t stream) {
    const float* x    = (const float*)d_in[0];   // [64,64,2048]
    const float* w    = (const float*)d_in[1];   // [128,64,5]
    const float* beta = (const float*)d_in[2];   // [1]
    const float* bvec = (const float*)d_in[3];   // [128]
    float* out = (float*)d_out;
    char*  ws  = (char*)d_ws;

    float*    conv    = (float*)   (ws);                 // 64 MB  [T][B][O]
    unsigned* sbits   = (unsigned*)(ws + 67108864);      //  2 MB  [B][O][64]
    float*    dmat    = (float*)   (ws + 69206016);      // 64 KB
    float*    invn    = (float*)   (ws + 69271552);      // 512 B
    half_t*   dBhi    = (half_t*)  (ws + 69272064);      // 32 KB
    half_t*   dBlo    = (half_t*)  (ws + 69304832);      // 32 KB
    half_t*   wBhi    = (half_t*)  (ws + 69337600);      // 128 KB
    half_t*   wBlo    = (half_t*)  (ws + 69468672);      // 128 KB
    unsigned* counter = (unsigned*)(ws + 69599744);      // 4 B

    k_zero  <<<1,    1,   0, stream>>>(counter);
    k_prep_d<<<64,   256, 0, stream>>>(w, dmat, invn);
    k_pack_dB<<<64,  256, 0, stream>>>(dmat, dBhi, dBlo);
    k_pack_wB<<<256, 256, 0, stream>>>(w, wBhi, wBlo);
    k_conv  <<<8192, 256, 0, stream>>>(x, wBhi, wBlo, conv);
    k_scan  <<<4,    256, 0, stream>>>(conv, dBhi, dBlo, invn, bvec, beta, sbits);
    k_expand<<<2048, 256, 0, stream>>>(sbits, out, counter);
    k_loss  <<<1,    1,   0, stream>>>(counter, out);
}
